// MultiHeadedAttentionCrossAtt_43138651521697
// MI455X (gfx1250) — compile-verified, matrix operand order checked by reading, not run on hardware
//
#include <hip/hip_runtime.h>

#define B_  8
#define T1_ 512
#define T2_ 1500
#define T2P 1504
#define NF  1024
#define NH  16
#define DK  64

typedef __attribute__((ext_vector_type(16))) __bf16    v16bf;
typedef __attribute__((ext_vector_type(8)))  float     v8f;
typedef __attribute__((ext_vector_type(4)))  unsigned  u32x4;

// types for the async-to-LDS builtin: V4i in AS(1) / AS(3)
typedef int v4i32 __attribute__((vector_size(16)));
typedef __attribute__((address_space(1))) v4i32 gv4i;
typedef __attribute__((address_space(3))) v4i32 lv4i;

union Frag {
    v16bf v;
    u32x4 q[2];
    unsigned short h[16];
};

__device__ inline unsigned short f2bf(float f) {
    unsigned u = __float_as_uint(f);
    u = u + 0x7FFFu + ((u >> 16) & 1u);   // round-to-nearest-even
    return (unsigned short)(u >> 16);
}

__device__ inline v8f wmma_bf16(v16bf a, v16bf b, v8f c) {
    return __builtin_amdgcn_wmma_f32_16x16x32_bf16(false, a, false, b, (short)0, c, false, false);
}

// 16-byte global -> LDS async copy (ASYNCcnt-tracked)
__device__ inline void async_b128(const void* gsrc, void* ldst) {
#if __has_builtin(__builtin_amdgcn_global_load_async_to_lds_b128)
    __builtin_amdgcn_global_load_async_to_lds_b128((gv4i*)gsrc, (lv4i*)ldst, 0, 0);
#else
    asm volatile("global_load_async_to_lds_b128 %0, %1, off" :: "v"((lv4i*)ldst), "v"((gv4i*)gsrc) : "memory");
#endif
}

__device__ inline void wait_async0() {
#if __has_builtin(__builtin_amdgcn_s_wait_asynccnt)
    __builtin_amdgcn_s_wait_asynccnt(0);
#else
    asm volatile("s_wait_asynccnt 0x0" ::: "memory");
#endif
}

// ---------------------------------------------------------------------------
// GEMM: C[M,N] = Ab[M,K](bf16) @ Wt[N,K](bf16, pre-transposed) + bias, f32 out.
// Block 256 threads (8 waves, 2x4). Block tile 64x128; wave tile 32x32 (2x2
// WMMA subtiles). K staged in double-buffered LDS via async-to-LDS copies.
// Requires N % 128 == 0, K % 32 == 0. M guarded.
// ---------------------------------------------------------------------------
__global__ __launch_bounds__(256) void gemm_kernel(
    const unsigned short* __restrict__ Ab, const unsigned short* __restrict__ Wt,
    const float* __restrict__ bias, float* __restrict__ C,
    int M, int N, int K)
{
    __shared__ __align__(16) unsigned short As[2][64 * 32];    //  8 KB
    __shared__ __align__(16) unsigned short Ws[2][128 * 32];   // 16 KB

    const int tid  = threadIdx.x;
    const int lane = tid & 31;
    const int wv   = tid >> 5;
    const int half = lane >> 4;
    const int l16  = lane & 15;
    const int wm = wv & 1;            // 0..1 -> M offset 32*wm
    const int wn = wv >> 1;           // 0..3 -> N offset 32*wn
    const size_t blockM = (size_t)blockIdx.y * 64;
    const size_t blockN = (size_t)blockIdx.x * 128;

    // async loader mapping: A tile 64x32 = 256 b128 chunks; W tile 128x32 = 512
    const int a_row = tid >> 2;
    const int a_cj  = (tid & 3) * 8;

    v8f acc[2][2];
    {
        v8f zero = {0,0,0,0,0,0,0,0};
        acc[0][0] = zero; acc[0][1] = zero; acc[1][0] = zero; acc[1][1] = zero;
    }

    const int nk = K / 32;

    // ---- issue tile loads for chunk 0 into buffer 0 ----
    {
        size_t ar = blockM + a_row; if (ar >= (size_t)M) ar = (size_t)M - 1;
        async_b128(Ab + ar * (size_t)K + a_cj, &As[0][a_row * 32 + a_cj]);
        #pragma unroll
        for (int i = 0; i < 2; ++i) {
            int c = tid + 256 * i;
            int wrow = c >> 2, wcj = (c & 3) * 8;
            async_b128(Wt + (blockN + wrow) * (size_t)K + wcj, &Ws[0][wrow * 32 + wcj]);
        }
    }

    for (int kc = 0; kc < nk; ++kc) {
        wait_async0();
        __syncthreads();

        const int cur = kc & 1;
        if (kc + 1 < nk) {
            const int nxt = cur ^ 1;
            const int k0 = (kc + 1) * 32;
            size_t ar = blockM + a_row; if (ar >= (size_t)M) ar = (size_t)M - 1;
            async_b128(Ab + ar * (size_t)K + k0 + a_cj, &As[nxt][a_row * 32 + a_cj]);
            #pragma unroll
            for (int i = 0; i < 2; ++i) {
                int c = tid + 256 * i;
                int wrow = c >> 2, wcj = (c & 3) * 8;
                async_b128(Wt + (blockN + wrow) * (size_t)K + k0 + wcj,
                           &Ws[nxt][wrow * 32 + wcj]);
            }
        }

        const int cb = 8 * half;   // per-lane-half K-chunk base
        Frag a0, a1, b0, b1;
        a0.q[0] = *(const u32x4*)&As[cur][(wm * 32 + l16) * 32 + cb];
        a0.q[1] = *(const u32x4*)&As[cur][(wm * 32 + l16) * 32 + cb + 16];
        a1.q[0] = *(const u32x4*)&As[cur][(wm * 32 + 16 + l16) * 32 + cb];
        a1.q[1] = *(const u32x4*)&As[cur][(wm * 32 + 16 + l16) * 32 + cb + 16];
        b0.q[0] = *(const u32x4*)&Ws[cur][(wn * 32 + l16) * 32 + cb];
        b0.q[1] = *(const u32x4*)&Ws[cur][(wn * 32 + l16) * 32 + cb + 16];
        b1.q[0] = *(const u32x4*)&Ws[cur][(wn * 32 + 16 + l16) * 32 + cb];
        b1.q[1] = *(const u32x4*)&Ws[cur][(wn * 32 + 16 + l16) * 32 + cb + 16];

        acc[0][0] = wmma_bf16(a0.v, b0.v, acc[0][0]);
        acc[0][1] = wmma_bf16(a0.v, b1.v, acc[0][1]);
        acc[1][0] = wmma_bf16(a1.v, b0.v, acc[1][0]);
        acc[1][1] = wmma_bf16(a1.v, b1.v, acc[1][1]);
    }

    // write-back: subtile (mi,ni): row = 32*wm + 16*mi + r + 8*half,
    //             col = 32*wn + 16*ni + l16
    #pragma unroll
    for (int mi = 0; mi < 2; ++mi) {
        #pragma unroll
        for (int r = 0; r < 8; ++r) {
            size_t row = blockM + wm * 32 + mi * 16 + 8 * half + r;
            if (row < (size_t)M) {
                #pragma unroll
                for (int ni = 0; ni < 2; ++ni) {
                    size_t col = blockN + wn * 32 + ni * 16 + l16;
                    C[row * (size_t)N + col] = acc[mi][ni][r] + bias[col];
                }
            }
        }
    }
}

// ---------------------------------------------------------------------------
// Elementwise f32 -> bf16
// ---------------------------------------------------------------------------
__global__ void cvt_bf16_kernel(const float* __restrict__ src,
                                unsigned short* __restrict__ dst, long n)
{
    long idx = (long)blockIdx.x * blockDim.x + threadIdx.x;
    if (idx < n) dst[idx] = f2bf(src[idx]);
}

// ---------------------------------------------------------------------------
// W[K,N] f32 -> Wt[N,K] bf16 (transpose)
// ---------------------------------------------------------------------------
__global__ void transpose_bf16_kernel(const float* __restrict__ W,
                                      unsigned short* __restrict__ Wt,
                                      int K, int N)
{
    long idx = (long)blockIdx.x * blockDim.x + threadIdx.x;
    if (idx >= (long)K * N) return;
    int k = (int)(idx / N);
    int n = (int)(idx % N);
    Wt[(size_t)n * K + k] = f2bf(W[idx]);
}

// ---------------------------------------------------------------------------
// Pack q (f32 [b,t1,h*dk]) -> bf16 [b,h,t1,dk], 1/sqrt(64) folded in.
// ---------------------------------------------------------------------------
__global__ void pack_q_kernel(const float* __restrict__ q, unsigned short* __restrict__ qb)
{
    size_t idx = (size_t)blockIdx.x * blockDim.x + threadIdx.x;
    if (idx >= (size_t)B_ * NH * T1_ * DK) return;
    int dk = (int)(idx & 63);
    int t1 = (int)((idx >> 6) & 511);
    int bh = (int)(idx >> 15);
    int b = bh >> 4, h = bh & 15;
    float v = q[((size_t)(b * T1_ + t1)) * NF + h * DK + dk] * 0.125f;
    qb[idx] = f2bf(v);
}

// ---------------------------------------------------------------------------
// Pack kv (f32 [b,t2,2*nf]) -> K bf16 [b,h,t2p,dk], V^T bf16 [b,h,dk,t2p]
// ---------------------------------------------------------------------------
__global__ void pack_kv_kernel(const float* __restrict__ kv,
                               unsigned short* __restrict__ kb,
                               unsigned short* __restrict__ vt)
{
    size_t idx = (size_t)blockIdx.x * blockDim.x + threadIdx.x;
    if (idx >= (size_t)B_ * NH * T2P * DK) return;
    int dk = (int)(idx & 63);
    size_t t = idx >> 6;
    int t2 = (int)(t % T2P);
    int bh = (int)(t / T2P);
    int b = bh >> 4, h = bh & 15;
    float kval = 0.f, vval = 0.f;
    if (t2 < T2_) {
        const float* row = kv + ((size_t)(b * T2_ + t2)) * (2 * NF);
        kval = row[h * DK + dk];
        vval = row[NF + h * DK + dk];
    }
    kb[((size_t)bh * T2P + t2) * DK + dk] = f2bf(kval);
    vt[((size_t)bh * DK + dk) * T2P + t2] = f2bf(vval);
}

// ---------------------------------------------------------------------------
// Flash attention: 1 wave per (b,h, 16-row Q tile); 32-key tiles;
// 4 WMMA QK^T + online softmax + LDS transpose + 4 WMMA P@V.
// ---------------------------------------------------------------------------
__global__ __launch_bounds__(32) void attn_kernel(
    const unsigned short* __restrict__ qb,
    const unsigned short* __restrict__ kb,
    const unsigned short* __restrict__ vt,
    const int* __restrict__ mask,
    float* __restrict__ ctx)
{
    __shared__ __align__(16) unsigned short pl[16 * 32];

    const int lane = threadIdx.x;
    const int half = lane >> 4;
    const int l16  = lane & 15;
    const int qt = blockIdx.x & 31;
    const int bh = blockIdx.x >> 5;
    const int b = bh >> 4;
    const int h = bh & 15;

    Frag qf[2];
    const unsigned short* qp = qb + ((size_t)bh * T1_ + qt * 16 + l16) * DK;
    #pragma unroll
    for (int w = 0; w < 2; ++w) {
        int base = 32 * w + 8 * half;
        qf[w].q[0] = *(const u32x4*)(qp + base);
        qf[w].q[1] = *(const u32x4*)(qp + base + 16);
    }

    v8f zero = {0,0,0,0,0,0,0,0};
    v8f cacc[4];
    #pragma unroll
    for (int i = 0; i < 4; ++i) cacc[i] = zero;
    float mrow[8], lrow[8];
    #pragma unroll
    for (int r = 0; r < 8; ++r) { mrow[r] = -1e30f; lrow[r] = 0.f; }

    const int* mp = mask + b * T2_;

    for (int kt = 0; kt < T2P / 32; ++kt) {
        const int k0 = kt * 32;

        v8f s[2];
        s[0] = zero; s[1] = zero;
        #pragma unroll
        for (int nt = 0; nt < 2; ++nt) {
            const unsigned short* kp = kb + ((size_t)bh * T2P + k0 + nt * 16 + l16) * DK;
            #pragma unroll
            for (int w = 0; w < 2; ++w) {
                int base = 32 * w + 8 * half;
                Frag kf;
                kf.q[0] = *(const u32x4*)(kp + base);
                kf.q[1] = *(const u32x4*)(kp + base + 16);
                s[nt] = wmma_bf16(qf[w].v, kf.v, s[nt]);
            }
        }

        #pragma unroll
        for (int nt = 0; nt < 2; ++nt) {
            int kg = k0 + nt * 16 + l16;
            bool valid = (kg < T2_) && (mp[kg < T2_ ? kg : 0] != 0);
            if (!valid) {
                #pragma unroll
                for (int r = 0; r < 8; ++r) s[nt][r] = -1e30f;
            }
        }

        float tmax[8];
        #pragma unroll
        for (int r = 0; r < 8; ++r) {
            float t = fmaxf(s[0][r], s[1][r]);
            t = fmaxf(t, __shfl_xor(t, 1, 32));
            t = fmaxf(t, __shfl_xor(t, 2, 32));
            t = fmaxf(t, __shfl_xor(t, 4, 32));
            t = fmaxf(t, __shfl_xor(t, 8, 32));
            tmax[r] = t;
        }
        float alpha[8];
        #pragma unroll
        for (int r = 0; r < 8; ++r) {
            float mn = fmaxf(mrow[r], tmax[r]);
            alpha[r] = __expf(mrow[r] - mn);
            mrow[r] = mn;
        }
        float psum[8];
        #pragma unroll
        for (int r = 0; r < 8; ++r) psum[r] = 0.f;
        #pragma unroll
        for (int nt = 0; nt < 2; ++nt) {
            #pragma unroll
            for (int r = 0; r < 8; ++r) {
                float sc = s[nt][r];
                float p = (sc <= -1e29f) ? 0.f : __expf(sc - mrow[r]);
                s[nt][r] = p;
                psum[r] += p;
            }
        }
        #pragma unroll
        for (int r = 0; r < 8; ++r) {
            float t = psum[r];
            t += __shfl_xor(t, 1, 32);
            t += __shfl_xor(t, 2, 32);
            t += __shfl_xor(t, 4, 32);
            t += __shfl_xor(t, 8, 32);
            lrow[r] = lrow[r] * alpha[r] + t;
        }
        #pragma unroll
        for (int i = 0; i < 4; ++i) {
            #pragma unroll
            for (int r = 0; r < 8; ++r) cacc[i][r] *= alpha[r];
        }

        __syncthreads();
        #pragma unroll
        for (int nt = 0; nt < 2; ++nt) {
            #pragma unroll
            for (int r = 0; r < 8; ++r)
                pl[(r + 8 * half) * 32 + nt * 16 + l16] = f2bf(s[nt][r]);
        }
        __syncthreads();
        Frag pf;
        pf.q[0] = *(const u32x4*)&pl[l16 * 32 + 8 * half];
        pf.q[1] = *(const u32x4*)&pl[l16 * 32 + 8 * half + 16];

        #pragma unroll
        for (int nt = 0; nt < 4; ++nt) {
            const unsigned short* vp =
                vt + ((size_t)bh * DK + nt * 16 + l16) * T2P + k0 + 8 * half;
            Frag vf;
            vf.q[0] = *(const u32x4*)(vp);
            vf.q[1] = *(const u32x4*)(vp + 16);
            cacc[nt] = wmma_bf16(pf.v, vf.v, cacc[nt]);
        }
    }

    #pragma unroll
    for (int r = 0; r < 8; ++r) lrow[r] = 1.f / fmaxf(lrow[r], 1e-30f);
    #pragma unroll
    for (int nt = 0; nt < 4; ++nt) {
        #pragma unroll
        for (int r = 0; r < 8; ++r) {
            int t1r = qt * 16 + r + 8 * half;
            int dk = nt * 16 + l16;
            ctx[((size_t)(b * T1_ + t1r)) * NF + h * DK + dk] = cacc[nt][r] * lrow[r];
        }
    }
}

// ---------------------------------------------------------------------------
extern "C" void kernel_launch(void* const* d_in, const int* in_sizes, int n_in,
                              void* d_out, int out_size, void* d_ws, size_t ws_size,
                              hipStream_t stream)
{
    const float* x    = (const float*)d_in[0];
    const float* mem  = (const float*)d_in[1];
    const int*   mmsk = (const int*)d_in[2];
    const float* Wq   = (const float*)d_in[3];
    const float* bq   = (const float*)d_in[4];
    const float* Wkv  = (const float*)d_in[5];
    const float* bkv  = (const float*)d_in[6];
    const float* Wout = (const float*)d_in[7];
    const float* bout = (const float*)d_in[8];
    float* out = (float*)d_out;

    char* ws = (char*)d_ws;
    size_t off = 0;
    unsigned short* xb    = (unsigned short*)(ws + off); off += (size_t)B_ * T1_ * NF * 2;       // 8.4 MB
    unsigned short* memb  = (unsigned short*)(ws + off); off += (size_t)B_ * T2_ * NF * 2;       // 24.6 MB
    unsigned short* Wqt   = (unsigned short*)(ws + off); off += (size_t)NF * NF * 2;             // 2.1 MB
    unsigned short* Wkvt  = (unsigned short*)(ws + off); off += (size_t)(2 * NF) * NF * 2;       // 4.2 MB
    unsigned short* Woutt = (unsigned short*)(ws + off); off += (size_t)NF * NF * 2;             // 2.1 MB
    float* qf  = (float*)(ws + off);                     off += (size_t)B_ * T1_ * NF * 4;       // 16.8 MB
    float* kvf = (float*)(ws + off);                     off += (size_t)B_ * T2_ * 2 * NF * 4;   // 98.3 MB
    unsigned short* qb = (unsigned short*)(ws + off);    off += (size_t)B_ * NH * T1_ * DK * 2;
    unsigned short* kb = (unsigned short*)(ws + off);    off += (size_t)B_ * NH * T2P * DK * 2;
    unsigned short* vt = (unsigned short*)(ws + off);    off += (size_t)B_ * NH * DK * T2P * 2;
    float* ctx = qf;                        // qf dead after pack_q
    unsigned short* ctxb = xb;              // xb dead after q-GEMM

    dim3 blk(256);

    // bf16 conversions / weight transposes
    long nx = (long)B_ * T1_ * NF;
    cvt_bf16_kernel<<<(unsigned)((nx + 255) / 256), blk, 0, stream>>>(x, xb, nx);
    long nm = (long)B_ * T2_ * NF;
    cvt_bf16_kernel<<<(unsigned)((nm + 255) / 256), blk, 0, stream>>>(mem, memb, nm);
    long nw1 = (long)NF * NF;
    transpose_bf16_kernel<<<(unsigned)((nw1 + 255) / 256), blk, 0, stream>>>(Wq, Wqt, NF, NF);
    long nw2 = (long)NF * 2 * NF;
    transpose_bf16_kernel<<<(unsigned)((nw2 + 255) / 256), blk, 0, stream>>>(Wkv, Wkvt, NF, 2 * NF);
    transpose_bf16_kernel<<<(unsigned)((nw1 + 255) / 256), blk, 0, stream>>>(Wout, Woutt, NF, NF);

    // q = x @ Wq + bq        (4096 x 1024 x 1024)
    gemm_kernel<<<dim3(NF / 128, (B_ * T1_) / 64), blk, 0, stream>>>(
        xb, Wqt, bq, qf, B_ * T1_, NF, NF);

    // kv = memory @ Wkv + bkv (12000 x 2048 x 1024)
    gemm_kernel<<<dim3((2 * NF) / 128, (B_ * T2_ + 63) / 64), blk, 0, stream>>>(
        memb, Wkvt, bkv, kvf, B_ * T2_, 2 * NF, NF);

    // operand packing for attention
    size_t nq = (size_t)B_ * NH * T1_ * DK;
    pack_q_kernel<<<(unsigned)((nq + 255) / 256), blk, 0, stream>>>(qf, qb);
    size_t nkv = (size_t)B_ * NH * T2P * DK;
    pack_kv_kernel<<<(unsigned)((nkv + 255) / 256), blk, 0, stream>>>(kvf, kb, vt);

    // flash attention
    attn_kernel<<<B_ * NH * (T1_ / 16), dim3(32), 0, stream>>>(qb, kb, vt, mmsk, ctx);

    // out = ctx @ Wout + bout (4096 x 1024 x 1024)
    cvt_bf16_kernel<<<(unsigned)((nx + 255) / 256), blk, 0, stream>>>(ctx, ctxb, nx);
    gemm_kernel<<<dim3(NF / 128, (B_ * T1_) / 64), blk, 0, stream>>>(
        ctxb, Woutt, bout, out, B_ * T1_, NF, NF);
}
